// Frequency_Aware_Delay_Attn_23192823398860
// MI455X (gfx1250) — matrix operand order repaired and checked by reading, not compiled, not run on hardware
//
#include <hip/hip_runtime.h>
#include <math.h>

// ---------------- problem constants ----------------
constexpr int B_  = 16;
constexpr int C_  = 4;
constexpr int N_  = 100;
constexpr int S_  = 96;
constexpr int L_  = 2 * S_;   // 192
constexpr int F_  = S_ + 1;   // 97
constexpr int FP_ = 128;      // F padded to multiple of 32
constexpr int D_  = 128;
constexpr int M_  = 16;
constexpr int H_  = 8;
constexpr int HD_ = D_ / H_;  // 16
constexpr int P_  = S_;       // 96
constexpr int R_  = B_ * C_ * N_;      // 6400 rows  (multiple of 64)
constexpr int RM_ = R_ * M_;           // 102400 rows (multiple of 64)
constexpr int NKEY_ = N_ * M_;         // 1600 keys per (b,c,h)
constexpr int KPL_  = NKEY_ / 32;      // 50 keys per lane (wave32)
constexpr float SCALE_ = 0.08838834764831845f;  // D^-0.5 (reference uses D)
constexpr float PI_ = 3.14159265358979323846f;

typedef __attribute__((ext_vector_type(16))) _Float16 v16h;
typedef __attribute__((ext_vector_type(8)))  _Float16 v8h;
typedef __attribute__((ext_vector_type(8)))  float    v8f;

// ---------------- WMMA fragment loaders: unconditional 128-bit loads ----------------
// A (16x32 f16), lane&15 = row, hi = lane>>4:
//   elements 0..7  -> K = k0 + 8*hi + e          (contiguous)
//   elements 8..15 -> K = k0 + 16 + 8*hi + (e-8) (contiguous)
__device__ __forceinline__ v16h frag_a(const _Float16* __restrict__ p /* row base + 8*hi */, int k0) {
  v8h lo = *(const v8h*)(p + k0);
  v8h hi = *(const v8h*)(p + k0 + 16);
  return __builtin_shufflevector(lo, hi, 0,1,2,3,4,5,6,7,8,9,10,11,12,13,14,15);
}
// B (32x16 f16), lane&15 = col, hi = lane>>4: elements 0..7 -> K = k0 + 8*hi + e, elements 8..15 -> K = k0 + 16 + 8*hi + (e-8)
__device__ __forceinline__ v16h frag_b(const _Float16* __restrict__ p /* col base + 8*hi */, int k0) {
  v8h lo = *(const v8h*)(p + k0);
  v8h hi = *(const v8h*)(p + k0 + 16);
  return __builtin_shufflevector(lo, hi, 0,1,2,3,4,5,6,7,8,9,10,11,12,13,14,15);
}

// ---------------- GEMM: Y = X1*W1^T [ + X2*W2^T ] + bias (+resid)(+relu) ----------------
// Any subtraction is pre-baked into negated packed weights, so the K-loop is pure load+WMMA.
// X*: f16 [rows, Kp] zero-padded; W*: f16 [>=NoutPad rows, Kp] zero-padded; Kp % 32 == 0.
// rows % 64 == 0, NoutPad % 16 == 0. Block = 128 threads = 4 waves; 16x16 tile per wave.
// Outputs: Yf (f32, ld=Nout, only cols < Nout) and/or Yh (f16, ld=NoutPad, pad cols get 0).
template<bool DUAL>
__global__ __launch_bounds__(128) void gemm_tile(
    const _Float16* __restrict__ X1, const _Float16* __restrict__ W1,
    const _Float16* __restrict__ X2, const _Float16* __restrict__ W2,
    const float* __restrict__ bias, const float* __restrict__ resid,
    float* __restrict__ Yf, _Float16* __restrict__ Yh,
    int Kp, int Nout, int NoutPad, int relu) {
  int lane = threadIdx.x & 31;
  int wav  = threadIdx.x >> 5;
  int rowBase = blockIdx.x * 64 + wav * 16;
  int colBase = blockIdx.y * 16;
  int lid = lane & 15;
  int hi  = lane >> 4;

  const _Float16* a1p = X1 + (size_t)(rowBase + lid) * Kp + 8 * hi;
  const _Float16* b1p = W1 + (size_t)(colBase + lid) * Kp + 8 * hi;
  const _Float16* a2p = DUAL ? (X2 + (size_t)(rowBase + lid) * Kp + 8 * hi) : nullptr;
  const _Float16* b2p = DUAL ? (W2 + (size_t)(colBase + lid) * Kp + 8 * hi) : nullptr;

  v8f acc = {};
  for (int k0 = 0; k0 < Kp; k0 += 32) {
    v16h a1 = frag_a(a1p, k0);
    v16h b1 = frag_b(b1p, k0);
    acc = __builtin_amdgcn_wmma_f32_16x16x32_f16(false, a1, false, b1, (short)0, acc, false, false);
    if (DUAL) {
      v16h a2 = frag_a(a2p, k0);
      v16h b2 = frag_b(b2p, k0);
      acc = __builtin_amdgcn_wmma_f32_16x16x32_f16(false, a2, false, b2, (short)0, acc, false, false);
    }
  }

  // C/D layout: lane&15 = col, element r -> row = rowBase + r + 8*hi
  int ncol = colBase + lid;
  bool incol = (ncol < Nout);
  float bv = (bias && incol) ? bias[ncol] : 0.0f;
#pragma unroll
  for (int r = 0; r < 8; ++r) {
    int m = rowBase + r + 8 * hi;                 // always in range (rows % 64 == 0)
    float v = acc[r] + bv;
    if (resid && incol) v += resid[(size_t)m * Nout + ncol];
    if (relu) v = fmaxf(v, 0.0f);
    if (Yf && incol) Yf[(size_t)m * Nout + ncol] = v;
    if (Yh) Yh[(size_t)m * NoutPad + ncol] = (_Float16)v;   // pad cols naturally produce 0
  }
}

// ---------------- pack scale * f32 [nr,nc] -> f16 [nrPad,ncPad], zero-padded ----------------
__global__ void pack_f16(const float* __restrict__ src, _Float16* __restrict__ dst,
                         int nr, int nc, int nrPad, int ncPad, float scale) {
  int i = blockIdx.x * blockDim.x + threadIdx.x;
  if (i >= nrPad * ncPad) return;
  int r = i / ncPad, c = i % ncPad;
  float v = (r < nr && c < nc) ? src[r * nc + c] * scale : 0.0f;
  dst[i] = (_Float16)v;
}

// ---------------- DFT / iDFT basis tables (built directly as padded f16) ----------------
// Forward  Cr[f*L+t] =  cos(2pi f t/L)*sig(f), Ci = -sin(...)*sig(f); rows padded 97->128
// Inverse  ICr[t*FP+f] = w_f cos(2pi f t/L)/L, ICi = -w_f sin(...)/L; cols padded 97->128
__global__ void build_tables(const float* __restrict__ fw,
                             _Float16* __restrict__ Cr, _Float16* __restrict__ Ci,
                             _Float16* __restrict__ ICr, _Float16* __restrict__ ICi) {
  int i = blockIdx.x * blockDim.x + threadIdx.x;
  if (i >= FP_ * L_) return;   // 128*192 == 192*128
  {
    int f = i / L_, t = i % L_;
    float cr = 0.0f, ci = 0.0f;
    if (f < F_) {
      float sig = 1.0f / (1.0f + __expf(-fw[f]));
      float th = 2.0f * PI_ * (float)((f * t) % L_) / (float)L_;
      cr =  cosf(th) * sig;
      ci = -sinf(th) * sig;
    }
    Cr[i] = (_Float16)cr;
    Ci[i] = (_Float16)ci;
  }
  {
    int t = i / FP_, f = i % FP_;
    float cr = 0.0f, ci = 0.0f;
    if (f < F_) {
      float w = (f == 0 || f == F_ - 1) ? 1.0f : 2.0f;
      float th = 2.0f * PI_ * (float)((f * t) % L_) / (float)L_;
      cr =  w * cosf(th) / (float)L_;
      ci = -w * sinf(th) / (float)L_;
    }
    ICr[i] = (_Float16)cr;
    ICi[i] = (_Float16)ci;
  }
}

// ---------------- xt = xf[...,None,:] * temp[M,F] -> padded f16 [RM, FP] ----------------
__global__ void xt_kernel(const float* __restrict__ xf_re, const float* __restrict__ xf_im,
                          const float* __restrict__ tr, const float* __restrict__ ti,
                          _Float16* __restrict__ xt_re, _Float16* __restrict__ xt_im, int total) {
  int i = blockIdx.x * blockDim.x + threadIdx.x;
  if (i >= total) return;            // total = RM * FP
  int f = i % FP_;
  int m = (i / FP_) % M_;
  int r = i / (FP_ * M_);
  float vr = 0.0f, vi = 0.0f;
  if (f < F_) {
    float ar = xf_re[r * F_ + f], ai = xf_im[r * F_ + f];
    float br = tr[m * F_ + f],    bi = ti[m * F_ + f];
    vr = ar * br - ai * bi;
    vi = ar * bi + ai * br;
  }
  xt_re[i] = (_Float16)vr;
  xt_im[i] = (_Float16)vi;
}

// ---------------- attention (per component): exact softmax, one wave per query row ----------------
// q:[R,D] f32, kproj/vproj:[RM,D] f32 in (bc,n,m) order; out: f16 [R,D] (feeds wo-GEMM).
__global__ __launch_bounds__(256) void attn_kernel(const float* __restrict__ q,
                                                   const float* __restrict__ kproj,
                                                   const float* __restrict__ vproj,
                                                   _Float16* __restrict__ out) {
  int lane = threadIdx.x & 31;
  int wav  = threadIdx.x >> 5;
  int row  = blockIdx.x * 8 + wav;            // 0 .. B*C*H*N-1 (51200, exact)
  int n  = row % N_;
  int h  = (row / N_) % H_;
  int bc = row / (N_ * H_);

  const float* qrow = q + ((size_t)(bc * N_ + n)) * D_ + h * HD_;
  float qv[HD_];
#pragma unroll
  for (int e = 0; e < HD_; ++e) qv[e] = qrow[e];

  const float* kbase = kproj + (size_t)bc * NKEY_ * D_ + h * HD_;
  const float* vbase = vproj + (size_t)bc * NKEY_ * D_ + h * HD_;

  float s[KPL_];
  float mx = -1e30f;
  for (int t = 0; t < KPL_; ++t) {
    const float* kr = kbase + (size_t)(lane + 32 * t) * D_;
    float acc = 0.0f;
#pragma unroll
    for (int e = 0; e < HD_; ++e) acc += qv[e] * kr[e];
    acc *= SCALE_;
    s[t] = acc;
    mx = fmaxf(mx, acc);
  }
  for (int msk = 16; msk >= 1; msk >>= 1) mx = fmaxf(mx, __shfl_xor(mx, msk, 32));

  float sum = 0.0f;
  for (int t = 0; t < KPL_; ++t) { s[t] = __expf(s[t] - mx); sum += s[t]; }
  for (int msk = 16; msk >= 1; msk >>= 1) sum += __shfl_xor(sum, msk, 32);
  float inv = 1.0f / sum;

  float oacc[HD_];
#pragma unroll
  for (int e = 0; e < HD_; ++e) oacc[e] = 0.0f;
  for (int t = 0; t < KPL_; ++t) {
    const float* vr = vbase + (size_t)(lane + 32 * t) * D_;
    float p = s[t] * inv;
#pragma unroll
    for (int e = 0; e < HD_; ++e) oacc[e] += p * vr[e];
  }

  float myout = 0.0f;
#pragma unroll
  for (int e = 0; e < HD_; ++e) {
    float rsum = oacc[e];
    for (int msk = 16; msk >= 1; msk >>= 1) rsum += __shfl_xor(rsum, msk, 32);
    if (lane == e) myout = rsum;
  }
  if (lane < HD_) out[((size_t)(bc * N_ + n)) * D_ + h * HD_ + lane] = (_Float16)myout;
}

// ---------------- host orchestration ----------------
static inline void launch_gemm(bool dual,
                               const _Float16* X1, const _Float16* W1,
                               const _Float16* X2, const _Float16* W2,
                               const float* bias, const float* resid,
                               float* Yf, _Float16* Yh,
                               int rows, int Kp, int Nout, int NoutPad, int relu,
                               hipStream_t stream) {
  dim3 grid(rows / 64, NoutPad / 16);
  if (dual)
    gemm_tile<true><<<grid, 128, 0, stream>>>(X1, W1, X2, W2, bias, resid, Yf, Yh, Kp, Nout, NoutPad, relu);
  else
    gemm_tile<false><<<grid, 128, 0, stream>>>(X1, W1, nullptr, nullptr, bias, resid, Yf, Yh, Kp, Nout, NoutPad, relu);
}

static inline void launch_pack(const float* src, _Float16* dst, int nr, int nc, int nrPad, int ncPad,
                               float scale, hipStream_t stream) {
  int tot = nrPad * ncPad;
  pack_f16<<<(tot + 255) / 256, 256, 0, stream>>>(src, dst, nr, nc, nrPad, ncPad, scale);
}

extern "C" void kernel_launch(void* const* d_in, const int* in_sizes, int n_in,
                              void* d_out, int out_size, void* d_ws, size_t ws_size,
                              hipStream_t stream) {
  (void)in_sizes; (void)n_in; (void)out_size; (void)ws_size;

  const float* xp      = (const float*)d_in[0];   // [R, L]
  const float* freq_w  = (const float*)d_in[1];
  const float* temp_r  = (const float*)d_in[2];
  const float* temp_i  = (const float*)d_in[3];
  const float* fc1_wr = (const float*)d_in[4];  const float* fc1_wi = (const float*)d_in[5];
  const float* fc1_br = (const float*)d_in[6];  const float* fc1_bi = (const float*)d_in[7];
  const float* fc2_wr = (const float*)d_in[8];  const float* fc2_wi = (const float*)d_in[9];
  const float* fc2_br = (const float*)d_in[10]; const float* fc2_bi = (const float*)d_in[11];
  const float* fc3_wr = (const float*)d_in[12]; const float* fc3_wi = (const float*)d_in[13];
  const float* fc3_br = (const float*)d_in[14]; const float* fc3_bi = (const float*)d_in[15];
  const float* fc4_wr = (const float*)d_in[16]; const float* fc4_wi = (const float*)d_in[17];
  const float* fc4_br = (const float*)d_in[18]; const float* fc4_bi = (const float*)d_in[19];
  const float* fc5_wr = (const float*)d_in[20]; const float* fc5_wi = (const float*)d_in[21];
  const float* fc5_br = (const float*)d_in[22]; const float* fc5_bi = (const float*)d_in[23];
  const float* ra_wq = (const float*)d_in[24]; const float* ra_bq = (const float*)d_in[25];
  const float* ra_wk = (const float*)d_in[26]; const float* ra_bk = (const float*)d_in[27];
  const float* ra_wv = (const float*)d_in[28]; const float* ra_bv = (const float*)d_in[29];
  const float* ra_wo = (const float*)d_in[30]; const float* ra_bo = (const float*)d_in[31];
  const float* ia_wq = (const float*)d_in[32]; const float* ia_bq = (const float*)d_in[33];
  const float* ia_wk = (const float*)d_in[34]; const float* ia_bk = (const float*)d_in[35];
  const float* ia_wv = (const float*)d_in[36]; const float* ia_bv = (const float*)d_in[37];
  const float* ia_wo = (const float*)d_in[38]; const float* ia_bo = (const float*)d_in[39];
  const float* op1_w = (const float*)d_in[40]; const float* op1_b = (const float*)d_in[41];
  const float* op2_w = (const float*)d_in[42]; const float* op2_b = (const float*)d_in[43];

  char* base = (char*)d_ws;
  size_t off = 0;
  auto allocF = [&](size_t n) { float* p = (float*)(base + off); off = (off + n * 4 + 15) & ~(size_t)15; return p; };
  auto allocH = [&](size_t n) { _Float16* p = (_Float16*)(base + off); off = (off + n * 2 + 15) & ~(size_t)15; return p; };

  // f16 padded buffers (GEMM operands)
  _Float16* xph    = allocH((size_t)R_ * L_);        // packed xp
  _Float16* Cr_h   = allocH((size_t)FP_ * L_);
  _Float16* Ci_h   = allocH((size_t)FP_ * L_);
  _Float16* ICr_h  = allocH((size_t)L_ * FP_);
  _Float16* ICi_h  = allocH((size_t)L_ * FP_);
  _Float16* xfh_re = allocH((size_t)R_ * FP_);
  _Float16* xfh_im = allocH((size_t)R_ * FP_);
  _Float16* xth_re = allocH((size_t)RM_ * FP_);
  _Float16* xth_im = allocH((size_t)RM_ * FP_);
  _Float16* e1h_re = allocH((size_t)RM_ * D_);
  _Float16* e1h_im = allocH((size_t)RM_ * D_);
  _Float16* e2h_re = allocH((size_t)RM_ * D_);
  _Float16* e2h_im = allocH((size_t)RM_ * D_);
  _Float16* g1h_re = allocH((size_t)R_ * D_);
  _Float16* g1h_im = allocH((size_t)R_ * D_);
  _Float16* gh_re  = allocH((size_t)R_ * D_);
  _Float16* gh_im  = allocH((size_t)R_ * D_);
  _Float16* aoh    = allocH((size_t)R_ * D_);
  _Float16* mhh_re = allocH((size_t)R_ * D_);
  _Float16* mhh_im = allocH((size_t)R_ * D_);
  _Float16* ofh_re = allocH((size_t)R_ * FP_);
  _Float16* ofh_im = allocH((size_t)R_ * FP_);
  _Float16* oth    = allocH((size_t)R_ * L_);
  _Float16* h1h    = allocH((size_t)R_ * 4 * P_);
  // packed weights (p* = +W, n* = -W for the complex-linear real parts)
  _Float16* pfc1_wr = allocH((size_t)D_ * FP_); _Float16* pfc1_wi = allocH((size_t)D_ * FP_);
  _Float16* nfc1_wi = allocH((size_t)D_ * FP_);
  _Float16* pfc2_wr = allocH((size_t)D_ * D_);  _Float16* pfc2_wi = allocH((size_t)D_ * D_);
  _Float16* nfc2_wi = allocH((size_t)D_ * D_);
  _Float16* pfc3_wr = allocH((size_t)FP_ * D_); _Float16* pfc3_wi = allocH((size_t)FP_ * D_);
  _Float16* nfc3_wi = allocH((size_t)FP_ * D_);
  _Float16* pfc4_wr = allocH((size_t)D_ * FP_); _Float16* pfc4_wi = allocH((size_t)D_ * FP_);
  _Float16* nfc4_wi = allocH((size_t)D_ * FP_);
  _Float16* pfc5_wr = allocH((size_t)D_ * D_);  _Float16* pfc5_wi = allocH((size_t)D_ * D_);
  _Float16* nfc5_wi = allocH((size_t)D_ * D_);
  _Float16* pra_wq = allocH((size_t)D_ * D_); _Float16* pra_wk = allocH((size_t)D_ * D_);
  _Float16* pra_wv = allocH((size_t)D_ * D_); _Float16* pra_wo = allocH((size_t)D_ * D_);
  _Float16* pia_wq = allocH((size_t)D_ * D_); _Float16* pia_wk = allocH((size_t)D_ * D_);
  _Float16* pia_wv = allocH((size_t)D_ * D_); _Float16* pia_wo = allocH((size_t)D_ * D_);
  _Float16* pop1   = allocH((size_t)4 * P_ * L_);
  _Float16* pop2   = allocH((size_t)P_ * 4 * P_);
  // f32 buffers (non-GEMM consumers)
  float* xf_re = allocF((size_t)R_ * F_);
  float* xf_im = allocF((size_t)R_ * F_);
  float* g_re  = allocF((size_t)R_ * D_);
  float* g_im  = allocF((size_t)R_ * D_);
  float* qp    = allocF((size_t)R_ * D_);
  float* kp    = allocF((size_t)RM_ * D_);
  float* vp    = allocF((size_t)RM_ * D_);

  // 0) pack inputs/weights to padded f16 (including negated wi copies)
  launch_pack(xp, xph, R_, L_, R_, L_, 1.f, stream);
  launch_pack(fc1_wr, pfc1_wr, D_, F_, D_, FP_, 1.f, stream);
  launch_pack(fc1_wi, pfc1_wi, D_, F_, D_, FP_, 1.f, stream);
  launch_pack(fc1_wi, nfc1_wi, D_, F_, D_, FP_, -1.f, stream);
  launch_pack(fc2_wr, pfc2_wr, D_, D_, D_, D_, 1.f, stream);
  launch_pack(fc2_wi, pfc2_wi, D_, D_, D_, D_, 1.f, stream);
  launch_pack(fc2_wi, nfc2_wi, D_, D_, D_, D_, -1.f, stream);
  launch_pack(fc3_wr, pfc3_wr, F_, D_, FP_, D_, 1.f, stream);
  launch_pack(fc3_wi, pfc3_wi, F_, D_, FP_, D_, 1.f, stream);
  launch_pack(fc3_wi, nfc3_wi, F_, D_, FP_, D_, -1.f, stream);
  launch_pack(fc4_wr, pfc4_wr, D_, F_, D_, FP_, 1.f, stream);
  launch_pack(fc4_wi, pfc4_wi, D_, F_, D_, FP_, 1.f, stream);
  launch_pack(fc4_wi, nfc4_wi, D_, F_, D_, FP_, -1.f, stream);
  launch_pack(fc5_wr, pfc5_wr, D_, D_, D_, D_, 1.f, stream);
  launch_pack(fc5_wi, pfc5_wi, D_, D_, D_, D_, 1.f, stream);
  launch_pack(fc5_wi, nfc5_wi, D_, D_, D_, D_, -1.f, stream);
  launch_pack(ra_wq, pra_wq, D_, D_, D_, D_, 1.f, stream);
  launch_pack(ra_wk, pra_wk, D_, D_, D_, D_, 1.f, stream);
  launch_pack(ra_wv, pra_wv, D_, D_, D_, D_, 1.f, stream);
  launch_pack(ra_wo, pra_wo, D_, D_, D_, D_, 1.f, stream);
  launch_pack(ia_wq, pia_wq, D_, D_, D_, D_, 1.f, stream);
  launch_pack(ia_wk, pia_wk, D_, D_, D_, D_, 1.f, stream);
  launch_pack(ia_wv, pia_wv, D_, D_, D_, D_, 1.f, stream);
  launch_pack(ia_wo, pia_wo, D_, D_, D_, D_, 1.f, stream);
  launch_pack(op1_w, pop1, 4 * P_, L_, 4 * P_, L_, 1.f, stream);
  launch_pack(op2_w, pop2, P_, 4 * P_, P_, 4 * P_, 1.f, stream);

  // 1) DFT / iDFT bases (sigmoid(freq_weights) folded into forward basis)
  build_tables<<<(FP_ * L_ + 255) / 256, 256, 0, stream>>>(freq_w, Cr_h, Ci_h, ICr_h, ICi_h);

  // 2) rfft as GEMM: xf = xp @ C^T  -> f32 [R,97] (for xt) + f16 [R,128] (for fc4)
  launch_gemm(false, xph, Cr_h, nullptr, nullptr, nullptr, nullptr, xf_re, xfh_re, R_, L_, F_, FP_, 0, stream);
  launch_gemm(false, xph, Ci_h, nullptr, nullptr, nullptr, nullptr, xf_im, xfh_im, R_, L_, F_, FP_, 0, stream);

  // 3) xt = xf * temp -> padded f16 [RM, 128]
  {
    int tot = RM_ * FP_;
    xt_kernel<<<(tot + 255) / 256, 256, 0, stream>>>(xf_re, xf_im, temp_r, temp_i, xth_re, xth_im, tot);
  }

  // 4) e = crelu(clin(xt, fc1)):  re = Xr*Wr - Xi*Wi (via -Wi), im = Xr*Wi + Xi*Wr
  launch_gemm(true, xth_re, pfc1_wr, xth_im, nfc1_wi, fc1_br, nullptr, nullptr, e1h_re, RM_, FP_, D_, D_, 1, stream);
  launch_gemm(true, xth_re, pfc1_wi, xth_im, pfc1_wr, fc1_bi, nullptr, nullptr, e1h_im, RM_, FP_, D_, D_, 1, stream);

  // 5) e = clin(e, fc2)
  launch_gemm(true, e1h_re, pfc2_wr, e1h_im, nfc2_wi, fc2_br, nullptr, nullptr, e2h_re, RM_, D_, D_, D_, 0, stream);
  launch_gemm(true, e1h_re, pfc2_wi, e1h_im, pfc2_wr, fc2_bi, nullptr, nullptr, e2h_im, RM_, D_, D_, D_, 0, stream);

  // 6) g = crelu(clin(xf, fc4))
  launch_gemm(true, xfh_re, pfc4_wr, xfh_im, nfc4_wi, fc4_br, nullptr, nullptr, g1h_re, R_, FP_, D_, D_, 1, stream);
  launch_gemm(true, xfh_re, pfc4_wi, xfh_im, pfc4_wr, fc4_bi, nullptr, nullptr, g1h_im, R_, FP_, D_, D_, 1, stream);

  // 7) g = clin(g, fc5) -> f32 (residual) + f16 (Q proj input)
  launch_gemm(true, g1h_re, pfc5_wr, g1h_im, nfc5_wi, fc5_br, nullptr, g_re, gh_re, R_, D_, D_, D_, 0, stream);
  launch_gemm(true, g1h_re, pfc5_wi, g1h_im, pfc5_wr, fc5_bi, nullptr, g_im, gh_im, R_, D_, D_, D_, 0, stream);

  // 8) real-part MHA: o_r = mha(g_re, e_re; ra_*) + g_re
  launch_gemm(false, gh_re,  pra_wq, nullptr, nullptr, ra_bq, nullptr, qp, nullptr, R_,  D_, D_, D_, 0, stream);
  launch_gemm(false, e2h_re, pra_wk, nullptr, nullptr, ra_bk, nullptr, kp, nullptr, RM_, D_, D_, D_, 0, stream);
  launch_gemm(false, e2h_re, pra_wv, nullptr, nullptr, ra_bv, nullptr, vp, nullptr, RM_, D_, D_, D_, 0, stream);
  attn_kernel<<<(B_ * C_ * H_ * N_) / 8, 256, 0, stream>>>(qp, kp, vp, aoh);
  launch_gemm(false, aoh, pra_wo, nullptr, nullptr, ra_bo, g_re, nullptr, mhh_re, R_, D_, D_, D_, 0, stream);

  // 9) imag-part MHA (reuses qp/kp/vp slabs)
  launch_gemm(false, gh_im,  pia_wq, nullptr, nullptr, ia_bq, nullptr, qp, nullptr, R_,  D_, D_, D_, 0, stream);
  launch_gemm(false, e2h_im, pia_wk, nullptr, nullptr, ia_bk, nullptr, kp, nullptr, RM_, D_, D_, D_, 0, stream);
  launch_gemm(false, e2h_im, pia_wv, nullptr, nullptr, ia_bv, nullptr, vp, nullptr, RM_, D_, D_, D_, 0, stream);
  attn_kernel<<<(B_ * C_ * H_ * N_) / 8, 256, 0, stream>>>(qp, kp, vp, aoh);
  launch_gemm(false, aoh, pia_wo, nullptr, nullptr, ia_bo, g_im, nullptr, mhh_im, R_, D_, D_, D_, 0, stream);

  // 10) o = clin(o, fc3)  [R,128] -> f16 [R,128p] (cols 97..127 zero)
  launch_gemm(true, mhh_re, pfc3_wr, mhh_im, nfc3_wi, fc3_br, nullptr, nullptr, ofh_re, R_, D_, F_, FP_, 0, stream);
  launch_gemm(true, mhh_re, pfc3_wi, mhh_im, pfc3_wr, fc3_bi, nullptr, nullptr, ofh_im, R_, D_, F_, FP_, 0, stream);

  // 11) irfft as GEMM: ot = of_re @ ICr^T + of_im @ ICi^T   [R,128p] -> f16 [R,192]
  launch_gemm(true, ofh_re, ICr_h, ofh_im, ICi_h, nullptr, nullptr, nullptr, oth, R_, FP_, L_, L_, 0, stream);

  // 12) out MLP: h = relu(ot @ op1^T + b1);  out = h @ op2^T + b2 -> d_out f32
  launch_gemm(false, oth, pop1, nullptr, nullptr, op1_b, nullptr, nullptr, h1h, R_, L_, 4 * P_, 4 * P_, 1, stream);
  launch_gemm(false, h1h, pop2, nullptr, nullptr, op2_b, nullptr, (float*)d_out, nullptr, R_, 4 * P_, P_, P_, 0, stream);
}
